// LSTMCell_73787538145444
// MI455X (gfx1250) — matrix operand order repaired and checked by reading, not compiled, not run on hardware
//
#include <hip/hip_runtime.h>

// ---------------------------------------------------------------------------
// LSTM cell, B=4096, I=H=1024, fp32 in/out.
//   gates = [x|hx] @ [w_ih|w_hh]^T + (b_ih + b_hh)   (M=4096, N=4096, K=2048)
// bf16 hi/lo split (3 passes: hi*hi + hi*lo + lo*hi) -> ~fp32 accuracy at
// v_wmma_f32_16x16x32_bf16 speed. Global->LDS via ASYNC copies (no staging
// VGPRs, no spills), triple-buffered, fused LSTM epilogue.
// Workspace: 64 MiB + 16 KiB.
// ---------------------------------------------------------------------------

typedef __attribute__((ext_vector_type(16))) __bf16 v16bf;
typedef __attribute__((ext_vector_type(8)))  float  v8f;
typedef __attribute__((ext_vector_type(4)))  int    v4i;

#define AS1 __attribute__((address_space(1)))
#define AS3 __attribute__((address_space(3)))

union Frag {
    v16bf f;
    uint4 q[2];
};

#define BDIM 4096
#define HDIM 1024
#define KDIM 2048

// LDS staging geometry: macro tile M=128, N=32 (x4 gates), K-slab 32
#define STR   40                     // padded row stride in bf16 (80B, 16B-aligned)
#define ABUF  (128 * STR)            // A slab: 128 rows
#define BUFSZ (ABUF + 128 * STR)     // + W slab: 4 gates x 32 cols = 128 rows
                                     // = 10240 elements = 20480 B per buffer
#define ITERS 192                    // 3 split terms * (2048/32) k-steps

__device__ __forceinline__ float sigmoid_f(float x) {
    return 1.0f / (1.0f + __expf(-x));
}

// ---- async global -> LDS 16-byte copy (per lane) ----
__device__ __forceinline__ void async_copy16(const __bf16* g, __bf16* l) {
#if __has_builtin(__builtin_amdgcn_global_load_async_to_lds_b128)
    __builtin_amdgcn_global_load_async_to_lds_b128(
        (AS1 v4i*)g, (AS3 v4i*)l, 0, 0);
#else
    unsigned loff = (unsigned)(unsigned long)(AS3 char*)l;
    asm volatile("global_load_async_to_lds_b128 %0, %1, off"
                 :: "v"(loff), "v"(g) : "memory");
#endif
}

__device__ __forceinline__ void wait_async_le8() {
#if __has_builtin(__builtin_amdgcn_s_wait_asynccnt)
    __builtin_amdgcn_s_wait_asynccnt(8);
#else
    asm volatile("s_wait_asynccnt 0x8" ::: "memory");
#endif
}

// ---- pack: fuse [src0|src1] rows (1024 each) into 2048-wide bf16 hi/lo ----
__global__ void pack_split_kernel(const float* __restrict__ src0,
                                  const float* __restrict__ src1,
                                  __bf16* __restrict__ hi,
                                  __bf16* __restrict__ lo) {
    int idx = blockIdx.x * 256 + threadIdx.x;        // < 4096*2048
    int row = idx >> 11;
    int k   = idx & 2047;
    float v = (k < 1024) ? src0[(row << 10) + k]
                         : src1[(row << 10) + (k - 1024)];
    __bf16 h = (__bf16)v;                            // RNE
    hi[idx] = h;
    lo[idx] = (__bf16)(v - (float)h);                // exact residual, then RNE
}

__global__ void bias_kernel(const float* __restrict__ a,
                            const float* __restrict__ b,
                            float* __restrict__ o) {
    int i = blockIdx.x * 256 + threadIdx.x;          // < 4096
    o[i] = a[i] + b[i];
}

// ---- fused GEMM + LSTM epilogue ----
__global__ __launch_bounds__(128)
void lstm_wmma_kernel(const __bf16* __restrict__ Ahi,
                      const __bf16* __restrict__ Alo,
                      const __bf16* __restrict__ Whi,
                      const __bf16* __restrict__ Wlo,
                      const float*  __restrict__ bias,
                      const float*  __restrict__ cx,
                      float*        __restrict__ out) {
    __shared__ __bf16 lds[3 * BUFSZ];                // 61440 B, triple buffered

    const int t    = threadIdx.x;                    // 0..127
    const int lane = t & 31;
    const int wm   = t >> 5;                         // wave id = 32-row slice
    const int l15  = lane & 15;
    const int lhi  = lane >> 4;

    const int n0   = blockIdx.x * 32;                // output cols (per gate)
    const int row0 = blockIdx.y * 128;               // batch rows

    v8f acc[4][2][2];                                // [gate][ntile][mtile]
#pragma unroll
    for (int g = 0; g < 4; ++g)
#pragma unroll
        for (int ni = 0; ni < 2; ++ni)
#pragma unroll
            for (int mi = 0; mi < 2; ++mi)
                acc[g][ni][mi] = (v8f){0.f, 0.f, 0.f, 0.f, 0.f, 0.f, 0.f, 0.f};

    // issue one K=32 slab (16 KB) as 8 async b128 ops per thread
    auto issue_slab = [&](int it, int buf) {
        const int term = it >> 6;                    // 0: hi*hi 1: hi*lo 2: lo*hi
        const int kk   = (it & 63) << 5;
        const __bf16* Ap = (term == 2) ? Alo : Ahi;
        const __bf16* Wp = (term == 1) ? Wlo : Whi;
        __bf16* L = &lds[buf * BUFSZ];
#pragma unroll
        for (int j = 0; j < 4; ++j) {                // A: 512 chunks / 128 thr
            int c = t + (j << 7), r = c >> 2, s = c & 3;
            async_copy16(Ap + (row0 + r) * KDIM + kk + s * 8,
                         L + r * STR + s * 8);
        }
#pragma unroll
        for (int j = 0; j < 4; ++j) {                // W: 512 chunks / 128 thr
            int c = t + (j << 7), r = c >> 2, s = c & 3;
            int grow = ((r >> 5) << 10) + n0 + (r & 31);  // gate*1024 + col
            async_copy16(Wp + grow * KDIM + kk + s * 8,
                         L + ABUF + r * STR + s * 8);
        }
    };

    auto compute = [&](int buf) {
        const __bf16* L = &lds[buf * BUFSZ];
        Frag a[2];
#pragma unroll
        for (int mi = 0; mi < 2; ++mi) {
            // A frag (16x32): lanes 0-15: K 0..7 / 16..23; lanes 16-31: +8
            int ad = (wm * 32 + mi * 16 + l15) * STR + lhi * 8;
            a[mi].q[0] = *(const uint4*)(L + ad);
            a[mi].q[1] = *(const uint4*)(L + ad + 16);
        }
#pragma unroll
        for (int g = 0; g < 4; ++g) {
#pragma unroll
            for (int ni = 0; ni < 2; ++ni) {
                // B frag (32x16): lane = col; lanes 0-15: K 0..7 / 16..23; lanes 16-31: +8
                int bd = ABUF + (g * 32 + ni * 16 + l15) * STR + lhi * 8;
                Frag b;
                b.q[0] = *(const uint4*)(L + bd);
                b.q[1] = *(const uint4*)(L + bd + 16);
#pragma unroll
                for (int mi = 0; mi < 2; ++mi)
                    acc[g][ni][mi] = __builtin_amdgcn_wmma_f32_16x16x32_bf16(
                        false, a[mi].f, false, b.f,
                        (short)0, acc[g][ni][mi], false, false);
            }
        }
    };

    // ---- triple-buffered async pipeline ----
    // slab i+2 writes buf (i+2)%3 == (i-1)%3, last read at iter i-1, and the
    // issue sits after the iter-i barrier -> race-free with 1 barrier/iter.
    issue_slab(0, 0);
    issue_slab(1, 1);
    for (int i = 0; i < ITERS; ++i) {
        wait_async_le8();                 // slab i complete (in-order returns)
        __syncthreads();                  // all waves' slab-i copies visible
        if (i + 2 < ITERS) issue_slab(i + 2, (i + 2) % 3);
        compute(i % 3);
    }

    // ---- fused LSTM epilogue ----
    float* hy1 = out;
    float* hy2 = out + (size_t)BDIM * HDIM;
    float* cyo = out + (size_t)2 * BDIM * HDIM;

#pragma unroll
    for (int ni = 0; ni < 2; ++ni) {
        const int col = n0 + ni * 16 + l15;
        const float bi = bias[col];
        const float bf = bias[col + 1024];
        const float bc = bias[col + 2048];
        const float bo = bias[col + 3072];
#pragma unroll
        for (int mi = 0; mi < 2; ++mi) {
            const int rowb = row0 + wm * 32 + mi * 16 + lhi * 8;
#pragma unroll
            for (int v = 0; v < 8; ++v) {
                const int row = rowb + v;
                const float ig = sigmoid_f(acc[0][ni][mi][v] + bi);
                const float fg = sigmoid_f(acc[1][ni][mi][v] + bf);
                const float cg = tanhf(acc[2][ni][mi][v] + bc);
                const float og = sigmoid_f(acc[3][ni][mi][v] + bo);
                const size_t o = (size_t)row * HDIM + col;
                const float cy = fg * cx[o] + ig * cg;
                const float hy = og * tanhf(cy);
                hy1[o] = hy;
                hy2[o] = hy;
                cyo[o] = cy;
            }
        }
    }
}

extern "C" void kernel_launch(void* const* d_in, const int* in_sizes, int n_in,
                              void* d_out, int out_size, void* d_ws, size_t ws_size,
                              hipStream_t stream) {
    const float* x    = (const float*)d_in[0];
    const float* hx   = (const float*)d_in[1];
    const float* cx   = (const float*)d_in[2];
    const float* w_ih = (const float*)d_in[3];
    const float* b_ih = (const float*)d_in[4];
    const float* w_hh = (const float*)d_in[5];
    const float* b_hh = (const float*)d_in[6];
    float* out = (float*)d_out;

    char* ws = (char*)d_ws;
    const size_t MATB = (size_t)4096 * 2048 * sizeof(__bf16);  // 16 MiB each
    __bf16* Ahi = (__bf16*)(ws);
    __bf16* Alo = (__bf16*)(ws + MATB);
    __bf16* Whi = (__bf16*)(ws + 2 * MATB);
    __bf16* Wlo = (__bf16*)(ws + 3 * MATB);
    float*  bsc = (float*)(ws + 4 * MATB);

    pack_split_kernel<<<32768, 256, 0, stream>>>(x, hx, Ahi, Alo);
    pack_split_kernel<<<32768, 256, 0, stream>>>(w_ih, w_hh, Whi, Wlo);
    bias_kernel<<<16, 256, 0, stream>>>(b_ih, b_hh, bsc);

    dim3 grid(HDIM / 32, BDIM / 128);   // (32, 32)
    lstm_wmma_kernel<<<grid, 128, 0, stream>>>(Ahi, Alo, Whi, Wlo, bsc, cx, out);
}
